// MultiTaskModel_12386685681716
// MI455X (gfx1250) — matrix operand order repaired and checked by reading, not compiled, not run on hardware
//
#include <hip/hip_runtime.h>
#include <hip/hip_bf16.h>

typedef __attribute__((ext_vector_type(16))) __bf16 v16bf;
typedef __attribute__((ext_vector_type(8)))  float  v8f;
typedef __attribute__((ext_vector_type(4)))  unsigned int v4u;
typedef __attribute__((vector_size(4 * sizeof(int)))) int v4ib;  // matches builtin prototype

union Frag  { v4u u[2]; v16bf v; };
union Pack8 { v4u u; __bf16 h[8]; };

#if defined(__gfx1250__) && __has_builtin(__builtin_amdgcn_global_load_async_to_lds_b128)
#define USE_ASYNC_B 1
#endif

__device__ __forceinline__ void wait_async0() {
#if defined(__gfx1250__)
#if __has_builtin(__builtin_amdgcn_s_wait_asynccnt)
  __builtin_amdgcn_s_wait_asynccnt(0);
#else
  asm volatile("s_wait_asynccnt 0x0" ::: "memory");
#endif
#endif
}

struct ConvP {
  const __bf16* __restrict__ in;   // activations, bf16, NCHW
  const __bf16* __restrict__ w;    // weights bf16, [(E,)Co, Ci*9] row-major (allocation padded!)
  const float*  __restrict__ bias; // f32 [(E,)Co]
  __bf16* __restrict__ out;        // bf16 NCHW
  const int* cls;                  // per-sample expert id, or nullptr
  int Ci, Hi, Wi, Co, Ho, Wo;
  int Ktot, stride, padLow, upsample, relu;
  int hwoSh, woSh, hwiSh, wiSh;    // log2 of Ho*Wo, Wo, Hi*Wi, Wi (all powers of two)
};

// Implicit-GEMM conv: M = B*Ho*Wo, N = Co, K = Ci*9.
// Block tile 64x64, 4 waves (wave32), each wave a 32x32 quadrant via 2x2 WMMA.
__global__ __launch_bounds__(128) void conv_wmma(ConvP p) {
  __shared__ __align__(16) unsigned char ldsA[64 * 80]; // 64 rows, 32 bf16 + pad (80B stride)
  __shared__ __align__(16) unsigned char ldsB[64 * 80]; // 64 cols (N), 32 bf16 K col-major + pad

  const int t    = threadIdx.x;
  const int m0   = blockIdx.x * 64;
  const int n0   = blockIdx.y * 64;
  const int lane = t & 31;
  const int wv   = t >> 5;
  const int wm   = (wv >> 1) * 32;
  const int wn   = (wv & 1) * 32;
  const int Wmask  = p.Wo - 1;
  const int HWmask = (1 << p.hwoSh) - 1;

  int e = 0;
  if (p.cls) e = p.cls[m0 >> p.hwoSh];       // block never spans samples when cls != null
  const long wOff = (long)e * p.Co * p.Ktot;
  const int  bOff = e * p.Co;

  v8f acc00 = {}, acc01 = {}, acc10 = {}, acc11 = {};

  // A-fill assignment: thread handles one K column (kkA) x 16 rows (rgA group)
  const int kkA = t & 31;
  const int rgA = t >> 5;
  const int mA  = m0 + rgA * 16;
  const int bA0  = mA >> p.hwoSh;
  const int rA   = mA & HWmask;
  const int oyA0 = rA >> p.woSh;
  const int oxA0 = rA & Wmask;

#if !defined(USE_ASYNC_B)
  const int clB = t & 63;
  const int khB = t >> 6;
#endif

  for (int k0 = 0; k0 < p.Ktot; k0 += 32) {
#if defined(USE_ASYNC_B)
    { // ---- B tile: async copy global->LDS, 256 x 16B chunks, 2 per thread ----
      const __bf16* wb = p.w + wOff + k0;
#pragma unroll
      for (int j = 0; j < 2; ++j) {
        const int c   = t * 2 + j;
        const int col = c >> 2;          // N column 0..63
        const int ch  = c & 3;           // 8-element K chunk 0..3
        const __bf16* g = wb + (long)(n0 + col) * p.Ktot + ch * 8;
        __builtin_amdgcn_global_load_async_to_lds_b128(
            (__attribute__((address_space(1))) v4ib*)g,
            (__attribute__((address_space(3))) v4ib*)(ldsB + col * 80 + ch * 16),
            0, 0);
      }
    }
#endif
    { // ---- A tile fill (im2col gather, zero-pad OOB / K tail) ----
      const int k = k0 + kkA;
      int ic = 0, ky = 0, kx = 0;
      const bool kOK = (k < p.Ktot);
      if (kOK) { ic = k / 9; const int r9 = k - ic * 9; ky = r9 / 3; kx = r9 - ky * 3; }
      int b = bA0, oy = oyA0, ox = oxA0;
#pragma unroll
      for (int i = 0; i < 16; ++i) {
        __bf16 v = (__bf16)0.0f;
        if (kOK) {
          int iy, ix; bool inb;
          if (p.upsample) { // nearest 2x upsample folded into gather (stride1, pad1)
            const int uy = oy + ky - 1, ux = ox + kx - 1;
            inb = ((unsigned)uy < (unsigned)(2 * p.Hi)) && ((unsigned)ux < (unsigned)(2 * p.Wi));
            iy = uy >> 1; ix = ux >> 1;
          } else {
            iy = oy * p.stride + ky - p.padLow;
            ix = ox * p.stride + kx - p.padLow;
            inb = ((unsigned)iy < (unsigned)p.Hi) && ((unsigned)ix < (unsigned)p.Wi);
          }
          if (inb) v = p.in[((long)(b * p.Ci + ic) << p.hwiSh) + (iy << p.wiSh) + ix];
        }
        *(__bf16*)(ldsA + (rgA * 16 + i) * 80 + kkA * 2) = v;
        if (++ox == p.Wo) { ox = 0; if (++oy == p.Ho) { oy = 0; ++b; } }
      }
    }
#if !defined(USE_ASYNC_B)
    { // ---- B tile fill (manual fallback; weights [Co, Ktot] row-major) ----
      const int n = n0 + clB;
      const bool nOK = (n < p.Co);
      const __bf16* wp = p.w + wOff + (long)n * p.Ktot + k0 + khB * 16;
#pragma unroll
      for (int i = 0; i < 16; ++i) {
        const int k = k0 + khB * 16 + i;
        __bf16 v = (__bf16)0.0f;
        if (nOK && k < p.Ktot) v = wp[i];
        *(__bf16*)(ldsB + clB * 80 + (khB * 16 + i) * 2) = v;
      }
    }
#else
    wait_async0();   // our wave's async B chunks are in LDS before we signal the barrier
#endif
    __syncthreads();
    { // ---- WMMA: fragment loads follow documented 16-bit A/B lane layouts ----
      const int r15 = lane & 15;
      const int hi  = lane >> 4;
      Frag a0, a1, b0, b1;
      const unsigned char* ar0 = ldsA + (wm + r15) * 80 + hi * 16;       // A: K0-7/K8-15 + K16-23/K24-31
      a0.u[0] = *(const v4u*)(ar0);
      a0.u[1] = *(const v4u*)(ar0 + 32);
      const unsigned char* ar1 = ldsA + (wm + 16 + r15) * 80 + hi * 16;
      a1.u[0] = *(const v4u*)(ar1);
      a1.u[1] = *(const v4u*)(ar1 + 32);
      const unsigned char* bc0 = ldsB + (wn + r15) * 80 + hi * 16;       // B: K0-7/K8-15 + K16-23/K24-31
      b0.u[0] = *(const v4u*)(bc0);
      b0.u[1] = *(const v4u*)(bc0 + 32);
      const unsigned char* bc1 = ldsB + (wn + 16 + r15) * 80 + hi * 16;
      b1.u[0] = *(const v4u*)(bc1);
      b1.u[1] = *(const v4u*)(bc1 + 32);
      acc00 = __builtin_amdgcn_wmma_f32_16x16x32_bf16(false, a0.v, false, b0.v, (short)0, acc00, false, false);
      acc01 = __builtin_amdgcn_wmma_f32_16x16x32_bf16(false, a0.v, false, b1.v, (short)0, acc01, false, false);
      acc10 = __builtin_amdgcn_wmma_f32_16x16x32_bf16(false, a1.v, false, b0.v, (short)0, acc10, false, false);
      acc11 = __builtin_amdgcn_wmma_f32_16x16x32_bf16(false, a1.v, false, b1.v, (short)0, acc11, false, false);
    }
    __syncthreads();
  }

  // ---- epilogue: each lane owns 8 consecutive M rows at fixed N -> 16B packed store ----
  const int r15 = lane & 15;
  const int hi8 = (lane >> 4) << 3;
  v8f accs[2][2] = {{acc00, acc01}, {acc10, acc11}};
#pragma unroll
  for (int ti = 0; ti < 2; ++ti) {
#pragma unroll
    for (int tj = 0; tj < 2; ++tj) {
      const int n = n0 + wn + tj * 16 + r15;
      if (n >= p.Co) continue;
      const long mBase = (long)m0 + wm + ti * 16 + hi8;
      const int  b    = (int)(mBase >> p.hwoSh);
      const long rem  = mBase & HWmask;
      const long oidx = (((long)b * p.Co + n) << p.hwoSh) + rem;  // 8-element aligned
      const float bv  = p.bias[bOff + n];
      Pack8 pk;
      const v8f a = accs[ti][tj];
#pragma unroll
      for (int i = 0; i < 8; ++i) {
        float v = a[i] + bv;
        if (p.relu) v = fmaxf(v, 0.0f);
        pk.h[i] = (__bf16)v;
      }
      *(v4u*)(p.out + oidx) = pk.u;
    }
  }
}

__global__ void cvt_f32_bf16(const float* __restrict__ in, __bf16* __restrict__ out, long n) {
  const long i = (long)blockIdx.x * blockDim.x + threadIdx.x;
  if (i < n) out[i] = (__bf16)in[i];
}

__global__ void pool_feats(const __bf16* __restrict__ feats, float* __restrict__ pooled) {
  const int i = blockIdx.x * blockDim.x + threadIdx.x; // b*512 + c
  if (i >= 64 * 512) return;
  const __bf16* f = feats + (long)i * 16;
  float s = 0.0f;
#pragma unroll
  for (int t = 0; t < 16; ++t) s += (float)f[t];
  pooled[i] = s * 0.0625f;
}

__global__ __launch_bounds__(128) void fc_cls(const float* __restrict__ pooled,
                                              const float* __restrict__ Wc1, const float* __restrict__ bc1,
                                              const float* __restrict__ Wc2, const float* __restrict__ bc2,
                                              float* __restrict__ logits, int* __restrict__ clsOut,
                                              int* __restrict__ clsWs) {
  const int b = blockIdx.x;
  const int j = threadIdx.x;
  __shared__ float hid[128];
  __shared__ float lg[5];
  const float* pb = pooled + b * 512;
  const float* wr = Wc1 + j * 512;
  float s = bc1[j];
  for (int k = 0; k < 512; ++k) s += pb[k] * wr[k];
  hid[j] = fmaxf(s, 0.0f);
  __syncthreads();
  if (j < 5) {
    const float* w2 = Wc2 + j * 128;
    float s2 = bc2[j];
    for (int k = 0; k < 128; ++k) s2 += hid[k] * w2[k];
    lg[j] = s2;
    logits[b * 5 + j] = s2;
  }
  __syncthreads();
  if (j == 0) {
    int best = 0; float bv = lg[0];
    for (int o = 1; o < 5; ++o) if (lg[o] > bv) { bv = lg[o]; best = o; } // first-max = jnp.argmax
    clsOut[b] = best;
    clsWs[b]  = best;
  }
}

// 1x1 head: [64,16,128,128] bf16 -> [64,5,128,128] f32, expert-routed weights (f32 direct).
__global__ void head1x1(const __bf16* __restrict__ in, const float* __restrict__ W6,
                        const float* __restrict__ b6, const int* __restrict__ cls,
                        float* __restrict__ seg) {
  const long pix = (long)blockIdx.x * blockDim.x + threadIdx.x;
  if (pix >= 64L * 16384) return;
  const int b = (int)(pix >> 14);
  const int ppos = (int)(pix & 16383);
  const int e = cls[b];
  const float* w  = W6 + e * 80;  // [5][16]
  const float* bi = b6 + e * 5;
  float v0 = bi[0], v1 = bi[1], v2 = bi[2], v3 = bi[3], v4 = bi[4];
  const __bf16* ib = in + (long)b * 16 * 16384 + ppos;
#pragma unroll
  for (int c = 0; c < 16; ++c) {
    const float xv = (float)ib[(long)c * 16384];
    v0 += xv * w[c]; v1 += xv * w[16 + c]; v2 += xv * w[32 + c];
    v3 += xv * w[48 + c]; v4 += xv * w[64 + c];
  }
  float* sp = seg + (long)b * 5 * 16384 + ppos;
  sp[0] = v0; sp[16384] = v1; sp[2 * 16384] = v2; sp[3 * 16384] = v3; sp[4 * 16384] = v4;
}

extern "C" void kernel_launch(void* const* d_in, const int* in_sizes, int n_in,
                              void* d_out, int out_size, void* d_ws, size_t ws_size,
                              hipStream_t stream) {
  const float* x = (const float*)d_in[0];
  const float* Wb[5] = {(const float*)d_in[1], (const float*)d_in[3], (const float*)d_in[5],
                        (const float*)d_in[7], (const float*)d_in[9]};
  const float* bb[5] = {(const float*)d_in[2], (const float*)d_in[4], (const float*)d_in[6],
                        (const float*)d_in[8], (const float*)d_in[10]};
  const float* Wc1 = (const float*)d_in[11]; const float* bc1 = (const float*)d_in[12];
  const float* Wc2 = (const float*)d_in[13]; const float* bc2 = (const float*)d_in[14];
  const float* We[6]; const float* be[6];
  for (int i = 0; i < 6; ++i) { We[i] = (const float*)d_in[15 + 2 * i]; be[i] = (const float*)d_in[16 + 2 * i]; }

  char* ws = (char*)d_ws;
  size_t cur = 0;
  auto alloc = [&](size_t bytes) -> char* {
    char* pp = ws + cur; cur += (bytes + 255) & ~(size_t)255; return pp;
  };

  __bf16* bufA  = (__bf16*)alloc(33554432);          // 64*64*64*64 bf16 (max activation)
  __bf16* bufB  = (__bf16*)alloc(33554432);
  __bf16* feats = (__bf16*)alloc(64L * 512 * 16 * 2);
  float*  pooled = (float*)alloc(64L * 512 * 4);
  int*    clsbuf = (int*)alloc(256);
  __bf16* xbf = bufB; // x_bf16 (6.3MB) aliases bufB; overwritten only after conv2 consumes conv1's bufA

  const long wbn[5] = {64L * 27, 128L * 576, 256L * 1152, 512L * 2304, 512L * 4608};
  const long wen[5] = {5L * 256 * 4608, 5L * 128 * 2304, 5L * 64 * 1152, 5L * 32 * 576, 5L * 16 * 288};
  __bf16* wbb[5]; __bf16* web[5];
  // +64KB slack per weight buffer: async B-tile copies may read N-tail/K-tail past the
  // logical end (those values are multiplied by zero-padded A columns or never stored).
  for (int i = 0; i < 5; ++i) wbb[i] = (__bf16*)alloc((size_t)wbn[i] * 2 + 65536);
  for (int i = 0; i < 5; ++i) web[i] = (__bf16*)alloc((size_t)wen[i] * 2 + 65536);

  auto cvt = [&](const float* src, __bf16* dst, long n) {
    cvt_f32_bf16<<<dim3((unsigned)((n + 255) / 256)), dim3(256), 0, stream>>>(src, dst, n);
  };
  cvt(x, xbf, 64L * 3 * 128 * 128);
  for (int i = 0; i < 5; ++i) cvt(Wb[i], wbb[i], wbn[i]);
  for (int i = 0; i < 5; ++i) cvt(We[i], web[i], wen[i]);

  auto lg2 = [](int v) { int s = 0; while ((1 << s) < v) ++s; return s; };

  auto conv = [&](const __bf16* in, const __bf16* w, const float* bias, __bf16* out,
                  const int* cls, int Ci, int Hi, int Wi, int Co, int Ho, int Wo,
                  int stride, int padLow, int ups, int relu) {
    ConvP p;
    p.in = in; p.w = w; p.bias = bias; p.out = out; p.cls = cls;
    p.Ci = Ci; p.Hi = Hi; p.Wi = Wi; p.Co = Co; p.Ho = Ho; p.Wo = Wo;
    p.Ktot = Ci * 9; p.stride = stride; p.padLow = padLow; p.upsample = ups; p.relu = relu;
    p.hwoSh = lg2(Ho * Wo); p.woSh = lg2(Wo); p.hwiSh = lg2(Hi * Wi); p.wiSh = lg2(Wi);
    dim3 grid((unsigned)((64L * Ho * Wo) / 64), (unsigned)((Co + 63) / 64));
    conv_wmma<<<grid, dim3(128), 0, stream>>>(p);
  };

  // Backbone (stride2, SAME => pad_low = 0 for even inputs)
  conv(xbf,  wbb[0], bb[0], bufA,  nullptr,   3, 128, 128,  64, 64, 64, 2, 0, 0, 1);
  conv(bufA, wbb[1], bb[1], bufB,  nullptr,  64,  64,  64, 128, 32, 32, 2, 0, 0, 1);
  conv(bufB, wbb[2], bb[2], bufA,  nullptr, 128,  32,  32, 256, 16, 16, 2, 0, 0, 1);
  conv(bufA, wbb[3], bb[3], bufB,  nullptr, 256,  16,  16, 512,  8,  8, 2, 0, 0, 1);
  conv(bufB, wbb[4], bb[4], feats, nullptr, 512,   8,   8, 512,  4,  4, 2, 0, 0, 1);

  // Classifier + routing
  pool_feats<<<dim3((64 * 512 + 255) / 256), dim3(256), 0, stream>>>(feats, pooled);
  fc_cls<<<dim3(64), dim3(128), 0, stream>>>(pooled, Wc1, bc1, Wc2, bc2,
                                             (float*)d_out, (int*)d_out + 320, clsbuf);

  // Expert decoder (upsample folded into gather; stride1 pad1)
  conv(feats, web[0], be[0], bufA, clsbuf, 512,  4,  4, 256,   8,   8, 1, 1, 1, 1);
  conv(bufA,  web[1], be[1], bufB, clsbuf, 256,  8,  8, 128,  16,  16, 1, 1, 1, 1);
  conv(bufB,  web[2], be[2], bufA, clsbuf, 128, 16, 16,  64,  32,  32, 1, 1, 1, 1);
  conv(bufA,  web[3], be[3], bufB, clsbuf,  64, 32, 32,  32,  64,  64, 1, 1, 1, 1);
  conv(bufB,  web[4], be[4], bufA, clsbuf,  32, 64, 64,  16, 128, 128, 1, 1, 1, 1);

  head1x1<<<dim3((unsigned)((64L * 16384 + 255) / 256)), dim3(256), 0, stream>>>(
      bufA, We[5], be[5], clsbuf, (float*)d_out + 384);
}
